// NNUE_86595130622448
// MI455X (gfx1250) — matrix operand order repaired and checked by reading, not compiled, not run on hardware
//
#include <hip/hip_runtime.h>

typedef __attribute__((ext_vector_type(16))) _Float16 v16h;
typedef __attribute__((ext_vector_type(8)))  _Float16 v8h;
typedef __attribute__((ext_vector_type(8)))  float    v8f;

#define HID    256
#define BAG    32
#define TILE_B 32      // positions per workgroup
#define XPAD   8       // f16 padding to stagger LDS banks (row stride 520*2=1040B)
#define APAD   8

__device__ __forceinline__ float clip01(float x) {
    return fminf(fmaxf(x, 0.0f), 1.0f);
}

// Load 16 consecutive f32 (64B-aligned) and convert to a v16h B-fragment slice.
__device__ __forceinline__ v16h cvt_row16(const float* __restrict__ wr) {
    const float4* q = (const float4*)wr;
    float4 a = q[0], b = q[1], c = q[2], d = q[3];
    v16h r;
    r[0]  = (_Float16)a.x; r[1]  = (_Float16)a.y; r[2]  = (_Float16)a.z; r[3]  = (_Float16)a.w;
    r[4]  = (_Float16)b.x; r[5]  = (_Float16)b.y; r[6]  = (_Float16)b.z; r[7]  = (_Float16)b.w;
    r[8]  = (_Float16)c.x; r[9]  = (_Float16)c.y; r[10] = (_Float16)c.z; r[11] = (_Float16)c.w;
    r[12] = (_Float16)d.x; r[13] = (_Float16)d.y; r[14] = (_Float16)d.z; r[15] = (_Float16)d.w;
    return r;
}

__device__ __forceinline__ v16h cat8(v8h lo, v8h hi) {
    return __builtin_shufflevector(lo, hi, 0,1,2,3,4,5,6,7,8,9,10,11,12,13,14,15);
}

__global__ __launch_bounds__(256) void nnue_fused(
    const int*   __restrict__ idx_us,
    const int*   __restrict__ idx_them,
    const float* __restrict__ emb,
    const float* __restrict__ l1w,    // [32][512]
    const float* __restrict__ l1b,    // [32]
    const float* __restrict__ l2w,    // [32][32]
    const float* __restrict__ l2b,    // [32]
    const float* __restrict__ outw,   // [1][32]
    const float* __restrict__ outb,   // [1]
    float*       __restrict__ out)    // [B]
{
    __shared__ __align__(32) _Float16 sx [TILE_B][2*HID + XPAD]; // ~33 KB activations (clipped, f16)
    __shared__ __align__(32) _Float16 sa1[TILE_B][32 + APAD];
    __shared__ __align__(32) _Float16 sa2[TILE_B][32 + APAD];

    const int tid   = threadIdx.x;
    const int lane  = tid & 31;
    const int wave  = tid >> 5;
    const int half  = (lane >> 4) & 1;   // which 16-lane half of the wave
    const int lane16 = lane & 15;
    const int bbase = blockIdx.x * TILE_B;

    // ---------- Phase 1: embedding-bag gather + sum + clip -> LDS f16 ----------
    // 64 (pos,perspective) bags, 8 per wave. Lane l holds bag-index l; broadcast
    // each via v_readlane so the 1KB row load has a scalar base + per-lane offset.
    for (int p = wave * 8; p < wave * 8 + 8; ++p) {
        const int pos  = p >> 1;
        const int prsp = p & 1;
        const int* ip  = (prsp ? idx_them : idx_us) + (size_t)(bbase + pos) * BAG;
        const int myidx = ip[lane];
        float acc[8] = {0.f,0.f,0.f,0.f,0.f,0.f,0.f,0.f};
        #pragma unroll 4
        for (int k = 0; k < BAG; ++k) {
            const int fi = __builtin_amdgcn_readlane(myidx, k);
            const float* row = emb + (size_t)fi * HID + lane * 8;
            const float4 r0 = ((const float4*)row)[0];
            const float4 r1 = ((const float4*)row)[1];
            acc[0] += r0.x; acc[1] += r0.y; acc[2] += r0.z; acc[3] += r0.w;
            acc[4] += r1.x; acc[5] += r1.y; acc[6] += r1.z; acc[7] += r1.w;
        }
        v8h h;
        #pragma unroll
        for (int t = 0; t < 8; ++t) h[t] = (_Float16)clip01(acc[t]);
        *(v8h*)&sx[pos][prsp * HID + lane * 8] = h;   // ds_store_b128
    }
    __syncthreads();

    // ---------- Phase 2: L1  (x[32x512] @ l1w.T[512x32]) via WMMA ----------
    // 4 output tiles (2 M-tiles x 2 N-tiles), one wave each, K=512 -> 16 WMMAs.
    if (wave < 4) {
        const int mt = wave >> 1, nt = wave & 1;
        const int mrow = mt * 16 + lane16;           // A: M = lane%16
        const int ncol = nt * 16 + lane16;           // B: N = lane%16
        v8f c = {};
        #pragma unroll
        for (int k0 = 0; k0 < 2 * HID; k0 += 32) {
            // A fragment: lane-half 0 -> K 0..15, half 1 -> K 16..31
            const v8h alo = *(const v8h*)&sx[mrow][k0 + half * 16];
            const v8h ahi = *(const v8h*)&sx[mrow][k0 + half * 16 + 8];
            const v16h a  = cat8(alo, ahi);
            // B fragment: lane-half 0 -> K k0..k0+15, half 1 -> K k0+16..k0+31,
            // contiguous along a row of l1w (row-major [32][512]).
            const v16h b  = cvt_row16(l1w + (size_t)ncol * (2 * HID) + k0 + half * 16);
            c = __builtin_amdgcn_wmma_f32_16x16x32_f16(false, a, false, b,
                                                       (short)0, c, false, false);
        }
        const float bias = l1b[ncol];
        #pragma unroll
        for (int j = 0; j < 8; ++j) {                // C: VGPR j -> M = j (+8 for half 1)
            const int m = mt * 16 + j + half * 8;
            sa1[m][ncol] = (_Float16)clip01(c[j] + bias);
        }
    }
    __syncthreads();

    // ---------- Phase 3: L2  (a1[32x32] @ l2w.T[32x32]), K=32 -> 1 WMMA/tile ----------
    if (wave < 4) {
        const int mt = wave >> 1, nt = wave & 1;
        const int mrow = mt * 16 + lane16;
        const int ncol = nt * 16 + lane16;
        const v8h alo = *(const v8h*)&sa1[mrow][half * 16];
        const v8h ahi = *(const v8h*)&sa1[mrow][half * 16 + 8];
        const v16h a  = cat8(alo, ahi);
        const v16h b  = cvt_row16(l2w + (size_t)ncol * 32 + half * 16);
        v8f c = {};
        c = __builtin_amdgcn_wmma_f32_16x16x32_f16(false, a, false, b,
                                                   (short)0, c, false, false);
        const float bias = l2b[ncol];
        #pragma unroll
        for (int j = 0; j < 8; ++j) {
            const int m = mt * 16 + j + half * 8;
            sa2[m][ncol] = (_Float16)clip01(c[j] + bias);
        }
    }
    __syncthreads();

    // ---------- Phase 4: output layer 32 -> 1 (wave 0, one position per lane) ----------
    if (wave == 0) {
        float acc = outb[0];
        #pragma unroll
        for (int f = 0; f < 32; ++f)
            acc += (float)sa2[lane][f] * outw[f];
        out[bbase + lane] = acc;
    }
}

extern "C" void kernel_launch(void* const* d_in, const int* in_sizes, int n_in,
                              void* d_out, int out_size, void* d_ws, size_t ws_size,
                              hipStream_t stream) {
    (void)n_in; (void)d_ws; (void)ws_size; (void)in_sizes;
    const int*   idx_us   = (const int*)  d_in[0];
    const int*   idx_them = (const int*)  d_in[1];
    const float* emb      = (const float*)d_in[2];
    const float* l1w      = (const float*)d_in[3];
    const float* l1b      = (const float*)d_in[4];
    const float* l2w      = (const float*)d_in[5];
    const float* l2b      = (const float*)d_in[6];
    const float* outw     = (const float*)d_in[7];
    const float* outb     = (const float*)d_in[8];
    float*       out      = (float*)d_out;

    const int batch  = out_size;               // [B,1] output
    const int blocks = batch / TILE_B;         // 131072/32 = 4096
    nnue_fused<<<blocks, 256, 0, stream>>>(idx_us, idx_them, emb,
                                           l1w, l1b, l2w, l2b, outw, outb, out);
}
